// CustomRNN_20804821581856
// MI455X (gfx1250) — matrix operand order repaired and checked by reading, not compiled, not run on hardware
//
#include <hip/hip_runtime.h>
#include <hip/hip_bf16.h>

// ---------------------------------------------------------------------------
// Dale RNN forward for MI455X (gfx1250).
//   * bf16 WMMA (v_wmma_f32_16x16x32_bf16) for r @ W_rec^T and r @ W_out^T
//   * f32 recurrent state kept in registers; only matmul operands quantized
//   * W_rec resident in LDS (132KB bf16, padded rows) -> uses CDNA5 320KB LDS
//   * noise streamed from HBM (512MB total = the roofline floor ~23us)
//   * drive (x@W_in^T + bias + noise) fused into the time loop, never stored
//   * z-matmul hoisted to one uniform branch per step (EXEC stays all-ones
//     across each WMMA region; minimal saveexec churn)
// Grid: 64 WGs x 256 threads (8 waves, wave32). Each WG owns 16 batch rows.
// ---------------------------------------------------------------------------

#define N_REC   256
#define N_IN    10
#define N_OUT   3
#define BATCH   1024
#define TDIM    512
#define ALPHA   0.2f
#define BM      16
#define WSTRIDE 264   // bf16 elems per padded row (256+8): 528B, 16B aligned

typedef __attribute__((ext_vector_type(16))) __bf16 v16bf;
typedef __attribute__((ext_vector_type(8)))  float  v8f;
typedef __attribute__((ext_vector_type(4)))  float  f32x4;

union BFVec { v16bf v; f32x4 q[2]; };

__device__ __forceinline__ v16bf ld_pair(const __bf16* p0, const __bf16* p1) {
  BFVec u;
  u.q[0] = *(const f32x4*)p0;   // ds_load_b128
  u.q[1] = *(const f32x4*)p1;   // ds_load_b128
  return u.v;
}

__global__ void zero_l2_kernel(float* out) {
  if (threadIdx.x == 0 && blockIdx.x == 0)
    out[(size_t)BATCH * TDIM * N_OUT] = 0.0f;
}

__global__ __launch_bounds__(256, 1)
void rnn_fused_kernel(const float* __restrict__ x,
                      const float* __restrict__ noise,
                      const float* __restrict__ W_in,
                      const float* __restrict__ W_rec,
                      const float* __restrict__ W_out_w,
                      const float* __restrict__ W_out_b,
                      const float* __restrict__ bias,
                      float* __restrict__ out)
{
  extern __shared__ __align__(16) char smem_raw[];
  __bf16* wlds    = (__bf16*)smem_raw;                 // [256][WSTRIDE] bf16 W_rec
  __bf16* rbf     = wlds  + 256 * WSTRIDE;             // [16][WSTRIDE]  bf16 r_{t-1}
  __bf16* wolds   = rbf   + 16  * WSTRIDE;             // [16][WSTRIDE]  bf16 W_out^T (padded)
  float*  winlds  = (float*)(wolds + 16 * WSTRIDE);    // [256][10]
  float*  biaslds = winlds + 256 * N_IN;               // [256]
  float*  xlds    = biaslds + 256;                     // [16][10]
  float*  l2lds   = xlds + BM * N_IN;                  // [1]

  const int tid  = threadIdx.x;
  const int lane = tid & 31;
  const int wave = tid >> 5;
  const int hl   = lane >> 4;     // lane-half (K/M group select per ISA layout)
  const int l16  = lane & 15;
  const int b0   = blockIdx.x * BM;

  // ---- one-time LDS fills (f32 -> bf16 conversion for matrix operands) ----
  for (int idx = tid; idx < 256 * 256; idx += 256)
    wlds[(idx >> 8) * WSTRIDE + (idx & 255)] = (__bf16)W_rec[idx];
  for (int idx = tid; idx < 16 * 256; idx += 256) {
    int n = idx >> 8, k = idx & 255;
    wolds[n * WSTRIDE + k] = (n < N_OUT) ? (__bf16)W_out_w[n * 256 + k]
                                         : (__bf16)0.0f;
  }
  for (int idx = tid; idx < 16 * WSTRIDE; idx += 256) rbf[idx] = (__bf16)0.0f;
  for (int idx = tid; idx < 256 * N_IN; idx += 256)   winlds[idx] = W_in[idx];
  biaslds[tid] = bias[tid];
  size_t xg = 0;
  if (tid < BM * N_IN) {
    xg = ((size_t)(b0 + tid / N_IN) * TDIM) * N_IN + (tid % N_IN);
    xlds[tid] = x[xg];                                  // x for t = 0
  }
  if (tid == 0) *l2lds = 0.0f;

  // ---- per-lane constant addressing ----
  // C/D layout: elem g of v8f <-> M = g + 8*hl, N = l16 (ISA 7.12.2)
  const int n0 = wave * 32 + l16;          // this wave's n-tiles: 2w, 2w+1
  const int n1 = n0 + 16;
  const float* nzp[8];
#pragma unroll
  for (int g = 0; g < 8; ++g)
    nzp[g] = noise + ((size_t)(b0 + g + hl * 8) * TDIM) * N_REC;

  // A frag: k = kc*32 + hl*16 + e (16 contiguous)
  const __bf16* Abase = rbf + l16 * WSTRIDE + hl * 16;
  // B frag: k = kc*32 + hl*16 + e (16 contiguous)          (same k order as A)
  const __bf16* Bbase0 = wlds + n0 * WSTRIDE + hl * 16;
  const __bf16* Bbase1 = wlds + n1 * WSTRIDE + hl * 16;
  const __bf16* Wbase  = wolds + l16 * WSTRIDE + hl * 16;  // W_out^T frags

  const int nout = l16;
  float wob = 0.0f;
  float* zp[8];
#pragma unroll
  for (int g = 0; g < 8; ++g) zp[g] = out;
  if (wave == 0 && nout < N_OUT) {
    wob = W_out_b[nout];
#pragma unroll
    for (int g = 0; g < 8; ++g)
      zp[g] = out + ((size_t)(b0 + g + hl * 8) * TDIM) * N_OUT + nout;
  }

  float rold[16];
#pragma unroll
  for (int s = 0; s < 16; ++s) rold[s] = 0.0f;
  float l2acc = 0.0f;

  __syncthreads();

  // =========================== time loop =================================
#pragma unroll 1
  for (int t = 0; t < TDIM; ++t) {
    // -- stream noise for this step (the HBM-bound part) + prefetch next --
    float nz[16];
#pragma unroll
    for (int s = 0; s < 16; ++s) {
      int g = s & 7;
      int n = (s < 8) ? n0 : n1;
      nz[s] = nzp[g][(size_t)t * N_REC + n];
    }
    if (t + 1 < TDIM) {
#pragma unroll
      for (int g = 0; g < 8; ++g)
        __builtin_prefetch(nzp[g] + (size_t)(t + 1) * N_REC + n0, 0, 1);
    }

    // -- fused drive: u = x_t @ W_in^T + bias + noise_t --
    float u[16];
#pragma unroll
    for (int s = 0; s < 16; ++s) {
      int g = s & 7;
      int m = g + hl * 8;
      int n = (s < 8) ? n0 : n1;
      float d = biaslds[n] + nz[s];
#pragma unroll
      for (int i = 0; i < N_IN; ++i)
        d = fmaf(xlds[m * N_IN + i], winlds[n * N_IN + i], d);
      u[s] = d;
    }

    // -- GEMM: r_{t-1} @ W_rec^T via bf16 WMMA, f32 accumulate --
    v8f acc0 = {}; v8f acc1 = {};
#pragma unroll
    for (int kc = 0; kc < 8; ++kc) {
      v16bf a  = ld_pair(Abase  + kc * 32, Abase  + kc * 32 + 8);
      v16bf bA = ld_pair(Bbase0 + kc * 32, Bbase0 + kc * 32 + 8);
      v16bf bB = ld_pair(Bbase1 + kc * 32, Bbase1 + kc * 32 + 8);
      acc0 = __builtin_amdgcn_wmma_f32_16x16x32_bf16(
                 false, a, false, bA, (short)0, acc0, false, false);
      acc1 = __builtin_amdgcn_wmma_f32_16x16x32_bf16(
                 false, a, false, bB, (short)0, acc1, false, false);
    }

    // -- wave 0: z_{t-1} = r_{t-1} @ W_out^T + b (one step behind; reads the
    //    same rbf staging). Single wave-uniform branch per step; EXEC is
    //    all-ones inside, as WMMA requires. --
    if (wave == 0) {
      v8f zacc = {};
#pragma unroll
      for (int kc = 0; kc < 8; ++kc) {
        v16bf a  = ld_pair(Abase + kc * 32, Abase + kc * 32 + 8);
        v16bf wb = ld_pair(Wbase + kc * 32, Wbase + kc * 32 + 8);
        zacc = __builtin_amdgcn_wmma_f32_16x16x32_bf16(
                   false, a, false, wb, (short)0, zacc, false, false);
      }
      if (t > 0 && nout < N_OUT) {
#pragma unroll
        for (int g = 0; g < 8; ++g)
          zp[g][(size_t)(t - 1) * N_OUT] = zacc[g] + wob;
      }
    }

    // -- state update (f32 master state lives in registers) --
#pragma unroll
    for (int s = 0; s < 16; ++s) {
      float pre = ((s < 8) ? acc0[s] : acc1[s - 8]) + u[s];
      float v = (1.0f - ALPHA) * rold[s] + ALPHA * fmaxf(pre, 0.0f);
      rold[s] = v;
      l2acc = fmaf(v, v, l2acc);
    }

    __syncthreads();   // all rbf reads (A frags, incl. wave0 z) complete
#pragma unroll
    for (int s = 0; s < 16; ++s) {
      int g = s & 7;
      int n = (s < 8) ? n0 : n1;
      rbf[(g + hl * 8) * WSTRIDE + n] = (__bf16)rold[s];
    }
    if (tid < BM * N_IN && t + 1 < TDIM)
      xlds[tid] = x[xg + (size_t)(t + 1) * N_IN];        // x for next step
    __syncthreads();
  }

  // -- final z_{T-1} from the last staged rbf --
  if (wave == 0) {
    v8f zacc = {};
#pragma unroll
    for (int kc = 0; kc < 8; ++kc) {
      v16bf a  = ld_pair(Abase + kc * 32, Abase + kc * 32 + 8);
      v16bf wb = ld_pair(Wbase + kc * 32, Wbase + kc * 32 + 8);
      zacc = __builtin_amdgcn_wmma_f32_16x16x32_bf16(
                 false, a, false, wb, (short)0, zacc, false, false);
    }
    if (nout < N_OUT) {
#pragma unroll
      for (int g = 0; g < 8; ++g)
        zp[g][(size_t)(TDIM - 1) * N_OUT] = zacc[g] + wob;
    }
  }

  // -- l2 reduction: LDS atomic, then one global atomic per WG --
  atomicAdd(l2lds, l2acc);
  __syncthreads();
  if (tid == 0) {
    const float scale = 1.0f / ((float)BATCH * (float)N_REC * (float)TDIM);
    atomicAdd(out + (size_t)BATCH * TDIM * N_OUT, (*l2lds) * scale);
  }
}

extern "C" void kernel_launch(void* const* d_in, const int* in_sizes, int n_in,
                              void* d_out, int out_size, void* d_ws, size_t ws_size,
                              hipStream_t stream) {
  (void)in_sizes; (void)n_in; (void)out_size; (void)d_ws; (void)ws_size;
  const float* x       = (const float*)d_in[0];
  const float* noise   = (const float*)d_in[1];
  const float* W_in    = (const float*)d_in[2];
  const float* W_rec   = (const float*)d_in[3];
  const float* W_out_w = (const float*)d_in[4];
  const float* W_out_b = (const float*)d_in[5];
  const float* bias    = (const float*)d_in[6];
  float* out = (float*)d_out;

  const size_t smem =
      (size_t)(256 + 16 + 16) * WSTRIDE * sizeof(__bf16) +   // W_rec, rbf, W_out^T
      (size_t)(256 * N_IN + 256 + BM * N_IN + 1) * sizeof(float);

  // >64KB dynamic LDS: raise the attribute (CDNA5 allows 320KB per WG).
  (void)hipFuncSetAttribute(reinterpret_cast<const void*>(rnn_fused_kernel),
                            hipFuncAttributeMaxDynamicSharedMemorySize,
                            (int)smem);

  zero_l2_kernel<<<1, 32, 0, stream>>>(out);
  rnn_fused_kernel<<<BATCH / BM, 256, smem, stream>>>(
      x, noise, W_in, W_rec, W_out_w, W_out_b, bias, out);
}
